// GraphAttentionLayer_4629974745121
// MI455X (gfx1250) — matrix operand order repaired and checked by reading, not compiled, not run on hardware
//
#include <hip/hip_runtime.h>
#include <hip/hip_bf16.h>
#include <stdint.h>

// ---- CDNA5 WMMA operand vector types -------------------------------------
typedef __attribute__((ext_vector_type(2)))  float  v2f;
typedef __attribute__((ext_vector_type(8)))  float  v8f;
typedef __attribute__((ext_vector_type(8)))  __bf16 v8bf;
typedef __attribute__((ext_vector_type(16))) __bf16 v16bf;

#define GAT_N     8192
#define GAT_INP   256
#define GAT_OUT   128
#define NEG_SLOPE 0.1f

__device__ __forceinline__ float leaky(float x) {
    return x > 0.0f ? x : NEG_SLOPE * x;
}

// ===========================================================================
// Kernel 1: Wh = h @ W_w   (fp32, V_WMMA_F32_16X16X4_F32)
//   grid = 512 blocks (16 rows each), block = 256 threads = 8 waves,
//   wave w owns output columns [16w, 16w+16).
//   Stores Wh fp32 (for si/sj) and WhbT bf16 transposed (B operand of kernel 3).
// ===========================================================================
__global__ __launch_bounds__(256)
void gat_wh_kernel(const float* __restrict__ h, const float* __restrict__ W,
                   float* __restrict__ Wh, __bf16* __restrict__ whbT) {
    const int lane = threadIdx.x & 31;
    const int wave = threadIdx.x >> 5;
    const int i0   = blockIdx.x * 16;
    const int m    = lane & 15;
    const int kb   = (lane < 16) ? 0 : 2;
    const int ncol = wave * 16 + m;

    const float* hrow = h + (size_t)(i0 + m) * GAT_INP + kb;
    const float* wcol = W + ncol;

    v8f acc = {};
    for (int k0 = 0; k0 < GAT_INP; k0 += 4) {
        v2f A = *(const v2f*)(hrow + k0);
        v2f B;
        B.x = wcol[(size_t)(k0 + kb)     * GAT_OUT];
        B.y = wcol[(size_t)(k0 + kb + 1) * GAT_OUT];
        acc = __builtin_amdgcn_wmma_f32_16x16x4_f32(
            false, A, false, B, (short)0, acc, false, false);
    }
#pragma unroll
    for (int r = 0; r < 8; ++r) {
        int mr  = (lane < 16) ? r : (8 + r);
        float v = acc[r];
        Wh[(size_t)(i0 + mr) * GAT_OUT + ncol]  = v;         // fp32 copy
        whbT[(size_t)ncol * GAT_N + (i0 + mr)]  = (__bf16)v; // transposed bf16
    }
}

// ===========================================================================
// Kernel 2: si = Wh @ a1, sj = Wh @ a2  (fp32 GEMV, one wave per row)
// ===========================================================================
__global__ __launch_bounds__(256)
void gat_scores_kernel(const float* __restrict__ Wh, const float* __restrict__ a_w,
                       float* __restrict__ si, float* __restrict__ sj) {
    const int lane = threadIdx.x & 31;
    const int wave = threadIdx.x >> 5;
    const int row  = blockIdx.x * 8 + wave;

    float4 v  = ((const float4*)(Wh + (size_t)row * GAT_OUT))[lane];
    float4 A1 = ((const float4*)a_w)[lane];
    float4 A2 = ((const float4*)(a_w + GAT_OUT))[lane];

    float p1 = v.x * A1.x + v.y * A1.y + v.z * A1.z + v.w * A1.w;
    float p2 = v.x * A2.x + v.y * A2.y + v.z * A2.z + v.w * A2.w;
#pragma unroll
    for (int off = 16; off > 0; off >>= 1) {
        p1 += __shfl_xor(p1, off, 32);
        p2 += __shfl_xor(p2, off, 32);
    }
    if (lane == 0) { si[row] = p1; sj[row] = p2; }
}

// ===========================================================================
// Kernel 3 (dominant): h_hat = leaky( leaky((si_i+sj_j)*adj_ij) @ Wh )
//   Fused, K=128 per barrier:
//     producer: each of 256 threads computes 8 scores (2x b128 adj loads,
//               8x cvt->bf16, one ds_store_b128) into double-buffered LDS
//     consumer: each of 8 waves issues 4x V_WMMA_F32_16X16X32_BF16 per
//               barrier; B operands (4x 32B contiguous from WhbT) are loaded
//               BEFORE the barrier to overlap the barrier wait.
//   adj streamed once from HBM (256 MB = the roofline floor);
//   WhbT (2 MB) stays L2-resident. LDS: 2 x 16x128 bf16 = 8 KB.
// ===========================================================================
__global__ __launch_bounds__(256)
void gat_attn_mm_kernel(const float* __restrict__ adj, const float* __restrict__ si,
                        const float* __restrict__ sj, const __bf16* __restrict__ whbT,
                        float* __restrict__ out) {
    __shared__ __bf16 att[2][16][128];   // 8 KB, double buffered

    const int tid  = threadIdx.x;
    const int lane = tid & 31;
    const int wave = tid >> 5;
    const int i0   = blockIdx.x * 16;

    // --- producer mapping: thread -> (row, 8-col group) of the 16x128 tile
    const int prow  = tid >> 4;            // 0..15
    const int pcolb = (tid & 15) * 8;      // 0,8,..,120
    const float si_r = si[i0 + prow];
    const float* adj_row = adj + (size_t)(i0 + prow) * GAT_N + pcolb;
    const float* sj_p    = sj + pcolb;

    // --- consumer mapping (bf16 A/B, same contiguous order k = 16*(lane/16) + s)
    const int cm    = lane & 15;
    const int koffA = (lane < 16) ? 0 : 16;   // A: K {koffA..koffA+15} contiguous
    const int joffB = (lane < 16) ? 0 : 16;   // B: lanes 16-31 take K+16
    const __bf16* bptr = whbT + (size_t)(wave * 16 + cm) * GAT_N + joffB;

    v8f acc = {};
    for (int j0 = 0; j0 < GAT_N; j0 += 128) {
        const int buf = (j0 >> 7) & 1;

        // ---- hoist B operand loads (independent of LDS -> overlap barrier)
        v16bf B0 = *(const v16bf*)(bptr + j0);
        v16bf B1 = *(const v16bf*)(bptr + j0 + 32);
        v16bf B2 = *(const v16bf*)(bptr + j0 + 64);
        v16bf B3 = *(const v16bf*)(bptr + j0 + 96);

        // ---- produce 8 score elements -> one ds_store_b128
        float4 a0 = *(const float4*)(adj_row + j0);
        float4 a1 = *(const float4*)(adj_row + j0 + 4);
        float4 z0 = *(const float4*)(sj_p + j0);
        float4 z1 = *(const float4*)(sj_p + j0 + 4);
        float av[8] = {a0.x, a0.y, a0.z, a0.w, a1.x, a1.y, a1.z, a1.w};
        float zv[8] = {z0.x, z0.y, z0.z, z0.w, z1.x, z1.y, z1.z, z1.w};
        v8bf pk;
#pragma unroll
        for (int e = 0; e < 8; ++e)
            pk[e] = (__bf16)leaky((si_r + zv[e]) * av[e]);
        *(v8bf*)&att[buf][prow][pcolb] = pk;

        if (j0 + 128 < GAT_N)
            __builtin_prefetch(adj_row + j0 + 128, 0, 0);  // global_prefetch_b8

        __syncthreads();

        // ---- consume: 4 back-to-back bf16 WMMAs (K = 128)
        v16bf Bk[4] = {B0, B1, B2, B3};
#pragma unroll
        for (int kk = 0; kk < 4; ++kk) {
            v8bf alo = *(const v8bf*)&att[buf][cm][kk * 32 + koffA];
            v8bf ahi = *(const v8bf*)&att[buf][cm][kk * 32 + koffA + 8];
            v16bf A  = __builtin_shufflevector(alo, ahi,
                         0,1,2,3,4,5,6,7,8,9,10,11,12,13,14,15);
            acc = __builtin_amdgcn_wmma_f32_16x16x32_bf16(
                false, A, false, Bk[kk], (short)0, acc, false, false);
        }
    }

#pragma unroll
    for (int r = 0; r < 8; ++r) {
        int mr = (lane < 16) ? r : (8 + r);
        out[(size_t)(i0 + mr) * GAT_OUT + wave * 16 + cm] = leaky(acc[r]);
    }
}

// ===========================================================================
extern "C" void kernel_launch(void* const* d_in, const int* in_sizes, int n_in,
                              void* d_out, int out_size, void* d_ws, size_t ws_size,
                              hipStream_t stream) {
    const float* h   = (const float*)d_in[0];   // [8192,256]
    const float* adj = (const float*)d_in[1];   // [8192,8192]
    const float* W   = (const float*)d_in[2];   // [256,128]
    const float* a_w = (const float*)d_in[3];   // [256]
    float* out = (float*)d_out;                 // [8192,128]

    // workspace: Wh fp32 (4 MB) | WhbT bf16 (2 MB) | si (32 KB) | sj (32 KB)
    char* ws = (char*)d_ws;
    float*  Wh   = (float*)ws;
    __bf16* whbT = (__bf16*)(ws + (size_t)GAT_N * GAT_OUT * sizeof(float));
    float*  si   = (float*)(ws + (size_t)GAT_N * GAT_OUT * (sizeof(float) + sizeof(__bf16)));
    float*  sj   = si + GAT_N;

    gat_wh_kernel     <<<GAT_N / 16, 256, 0, stream>>>(h, W, Wh, whbT);
    gat_scores_kernel <<<GAT_N / 8,  256, 0, stream>>>(Wh, a_w, si, sj);
    gat_attn_mm_kernel<<<GAT_N / 16, 256, 0, stream>>>(adj, si, sj, whbT, out);
}
